// FrequencyAwareAGv3_32263794327653
// MI455X (gfx1250) — matrix operand order repaired and checked by reading, not compiled, not run on hardware
//
#include <hip/hip_runtime.h>
#include <hip/hip_bf16.h>

// FrequencyAwareAG for MI455X (gfx1250, wave32, WMMA).
//
// Pipeline:
//   0) build orthonormal DCT-II matrix D (64x64) + D^T on device; Wc -> bf16^T
//   1) gmap = g @ Wg + bg                           (32x512, tiny)
//   2) buf1 = DCT along W :  per (n,h):  D  @ X     (f32 WMMA 16x16x4)
//   3) buf2 = DCT along H :  per (n,k2): D  @ buf1  (f32 WMMA 16x16x4)
//   4) buf1 = sigmoid(buf2 @ Wc + bc + gmap) * buf2 (bf16 WMMA 16x16x32, f32 acc)
//   5) buf2 = IDCT along H:  per (n,k2): D^T @ buf1 (f32 WMMA)
//   6) out  = x + IDCT along W: per (n,h): D^T @ buf2 (f32 WMMA, fused residual)
//
// Roofline: DCT/IDCT = 34 GFLOP f32; conv = 69 GFLOP on the bf16 pipe (8x the
// FLOP/inst of f32 K=4). Streamed bytes ~2.7 GB -> ~115 us @ 23.3 TB/s, so the
// pipeline is HBM-bound once the conv is off the K=4 f32 shape. The bf16 drop
// only affects the sigmoid-gate logits; x_freq, DCTs and residual stay f32.

typedef __attribute__((ext_vector_type(2)))  float  v2f;
typedef __attribute__((ext_vector_type(4)))  float  v4f;
typedef __attribute__((ext_vector_type(8)))  float  v8f;
typedef __attribute__((ext_vector_type(16))) __bf16 v16bf;
typedef __attribute__((ext_vector_type(8)))  __bf16 v8bf;
typedef __attribute__((ext_vector_type(4)))  __bf16 v4bf;

__device__ __forceinline__ v8f wmma4(v2f a, v2f b, v8f c) {
  return __builtin_amdgcn_wmma_f32_16x16x4_f32(
      /*neg_a=*/false, a, /*neg_b=*/false, b,
      /*c_mod=*/(short)0, c, /*reuse_a=*/false, /*reuse_b=*/false);
}

__device__ __forceinline__ v8f wmma_bf16(v16bf a, v16bf b, v8f c) {
  return __builtin_amdgcn_wmma_f32_16x16x32_bf16(
      /*neg_a=*/false, a, /*neg_b=*/false, b,
      /*c_mod=*/(short)0, c, /*reuse_a=*/false, /*reuse_b=*/false);
}

// ---------------------------------------------------------------------------
// Build DCT matrix: D[k][m] = s(k) * cos(pi*(m+0.5)*k/64), Dt = D^T
// ---------------------------------------------------------------------------
__global__ __launch_bounds__(256) void init_dct_kernel(float* __restrict__ D,
                                                       float* __restrict__ Dt) {
  int idx = blockIdx.x * 256 + threadIdx.x;     // 0..4095
  if (idx >= 4096) return;
  int k = idx >> 6, m = idx & 63;
  float scale = (k == 0) ? 0.125f : 0.17677669529663689f;  // sqrt(1/64), sqrt(2/64)
  float v = scale * __cosf(3.14159265358979323846f * ((float)m + 0.5f) * (float)k / 64.0f);
  D[k * 64 + m] = v;
  Dt[m * 64 + k] = v;
}

// ---------------------------------------------------------------------------
// WcT[d][k] = (bf16) Wc[k][d]   (512x512)
// ---------------------------------------------------------------------------
__global__ __launch_bounds__(256) void convert_wc_kernel(const float* __restrict__ Wc,
                                                         __bf16* __restrict__ wcT) {
  int idx = blockIdx.x * 256 + threadIdx.x;     // 0..262143
  int k = idx >> 9, d = idx & 511;
  wcT[(long)d * 512 + k] = (__bf16)Wc[idx];
}

// ---------------------------------------------------------------------------
// gmap[n][d] = sum_k g[n][k] * Wg[k][d] + bg[d]   (32 x 512, K=512) — tiny
// ---------------------------------------------------------------------------
__global__ __launch_bounds__(256) void gate_gemm_kernel(const float* __restrict__ g,
                                                        const float* __restrict__ Wg,
                                                        const float* __restrict__ bg,
                                                        float* __restrict__ gmap) {
  int idx = blockIdx.x * 256 + threadIdx.x;     // 0..16383
  int n = idx >> 9, d = idx & 511;
  float s = bg[d];
  for (int k = 0; k < 512; ++k) s += g[n * 512 + k] * Wg[k * 512 + d];
  gmap[idx] = s;
}

// ---------------------------------------------------------------------------
// Generic batched DCT GEMM:  O = A(64x64) @ B(64x512, strided) [+ resid]
// batch b -> (n = b>>6, j = b&63); base = n*2097152 + j*jStride
// B/O row r at base + r*rowStride + col. 8 waves, each owns 64 output cols.
// A staged in LDS (16 KB); per wave 4x4 C-tiles, 16 K-steps of K=4 f32 WMMA.
// ---------------------------------------------------------------------------
__global__ __launch_bounds__(256) void dct_gemm_kernel(const float* __restrict__ A,
                                                       const float* __restrict__ B,
                                                       const float* __restrict__ resid,
                                                       float* __restrict__ O,
                                                       long jStride, long rowStride) {
  __shared__ float As[64 * 64];
  const int tid = threadIdx.x;
  for (int i = tid; i < 4096; i += 256) As[i] = A[i];
  __syncthreads();

  const int b = blockIdx.x;
  const int n = b >> 6, j = b & 63;
  const long base = (long)n * 2097152L + (long)j * jStride;
  const float* __restrict__ Bp = B + base;
  float* __restrict__ Op = O + base;
  const float* __restrict__ Rp = resid ? (resid + base) : nullptr;

  const int wave = tid >> 5;
  const int lane = tid & 31;
  const int l16 = lane & 15;
  const int hi = lane >> 4;
  const int colBase = wave * 64;

  v8f acc[4][4];
#pragma unroll
  for (int mt = 0; mt < 4; ++mt)
#pragma unroll
    for (int nt = 0; nt < 4; ++nt) acc[mt][nt] = (v8f){0, 0, 0, 0, 0, 0, 0, 0};

  for (int ks = 0; ks < 16; ++ks) {
    const int k0 = ks * 4 + hi * 2;    // lanes 0-15 -> K {0,1}; 16-31 -> K {2,3}
    v2f a[4];
#pragma unroll
    for (int mt = 0; mt < 4; ++mt) {
      const int row = mt * 16 + l16;
      a[mt].x = As[row * 64 + k0];
      a[mt].y = As[row * 64 + k0 + 1];
    }
#pragma unroll
    for (int nt = 0; nt < 4; ++nt) {
      const int col = colBase + nt * 16 + l16;
      v2f bb;
      bb.x = Bp[(long)(k0) * rowStride + col];
      bb.y = Bp[(long)(k0 + 1) * rowStride + col];
#pragma unroll
      for (int mt = 0; mt < 4; ++mt) acc[mt][nt] = wmma4(a[mt], bb, acc[mt][nt]);
    }
  }

  // C/D layout: VGPR i -> rows (i, i+8) per lane half; col = lane&15.
#pragma unroll
  for (int mt = 0; mt < 4; ++mt) {
#pragma unroll
    for (int nt = 0; nt < 4; ++nt) {
      const int col = colBase + nt * 16 + l16;
#pragma unroll
      for (int i = 0; i < 8; ++i) {
        const int row = mt * 16 + hi * 8 + i;
        const long idx = (long)row * rowStride + col;
        float v = acc[mt][nt][i];
        if (Rp) v += Rp[idx];
        Op[idx] = v;
      }
    }
  }
}

// ---------------------------------------------------------------------------
// Frequency-domain 1x1 conv + gate (bf16 WMMA, f32 accumulate):
//   gated[m][d] = xf[m][d] * sigmoid( (xf @ Wc)[m][d] + bc[d] + gmap[m>>12][d] )
// M = 131072 rows, K = N = 512. Block: 64 rows x 256 cols; 8 waves of 64x32.
// A tile (64x512) converted to bf16 into LDS once per block (64 KB), then
// 16 K-steps of V_WMMA_F32_16X16X32_BF16 per C-tile.
// ---------------------------------------------------------------------------
__global__ __launch_bounds__(256) void conv_gate_kernel(const float* __restrict__ xf,
                                                        const __bf16* __restrict__ wcT,
                                                        const float* __restrict__ bc,
                                                        const float* __restrict__ gmap,
                                                        float* __restrict__ gated) {
  __shared__ __bf16 Abf[64 * 512];   // 64 KB
  const int tid = threadIdx.x;
  const int mBase = blockIdx.x * 64;
  const long aBase = (long)mBase * 512;

  // cooperative f32 -> bf16 staging of the activation tile
  for (int i = tid; i < 8192; i += 256) {
    v4f v = *reinterpret_cast<const v4f*>(xf + aBase + (long)i * 4);
    v4bf w;
    w[0] = (__bf16)v[0]; w[1] = (__bf16)v[1];
    w[2] = (__bf16)v[2]; w[3] = (__bf16)v[3];
    *reinterpret_cast<v4bf*>(Abf + i * 4) = w;
  }
  __syncthreads();

  const int wave = tid >> 5;
  const int lane = tid & 31;
  const int l16 = lane & 15;
  const int hi = lane >> 4;
  const int nBase = blockIdx.y * 256 + wave * 32;
  const int nIdx = mBase >> 12;      // image index (64-row blocks never straddle)

  v8f acc[4][2];
#pragma unroll
  for (int mt = 0; mt < 4; ++mt)
#pragma unroll
    for (int nt = 0; nt < 2; ++nt) acc[mt][nt] = (v8f){0, 0, 0, 0, 0, 0, 0, 0};

  for (int ks = 0; ks < 16; ++ks) {
    const int kb = ks * 32;
    // A frag (16x32 bf16): elem e -> K = kb+hi*16+e (contiguous)
    v16bf a[4];
#pragma unroll
    for (int mt = 0; mt < 4; ++mt) {
      const __bf16* lp = Abf + (mt * 16 + l16) * 512 + kb + hi * 16;
      v8bf a0 = *reinterpret_cast<const v8bf*>(lp);        // ds_load_b128
      v8bf a1 = *reinterpret_cast<const v8bf*>(lp + 8);    // ds_load_b128
#pragma unroll
      for (int e = 0; e < 8; ++e) { a[mt][e] = a0[e]; a[mt][8 + e] = a1[e]; }
    }
#pragma unroll
    for (int nt = 0; nt < 2; ++nt) {
      // B frag (32x16 bf16): per-lane K = kb + hi*16 + e, contiguous in WcT
      const __bf16* bp = wcT + (long)(nBase + nt * 16 + l16) * 512 + kb + hi * 16;
      v16bf b = *reinterpret_cast<const v16bf*>(bp);       // 2x b128
#pragma unroll
      for (int mt = 0; mt < 4; ++mt) acc[mt][nt] = wmma_bf16(a[mt], b, acc[mt][nt]);
    }
  }

#pragma unroll
  for (int mt = 0; mt < 4; ++mt) {
#pragma unroll
    for (int nt = 0; nt < 2; ++nt) {
      const int d = nBase + nt * 16 + l16;
      const float bias = bc[d] + gmap[nIdx * 512 + d];
#pragma unroll
      for (int i = 0; i < 8; ++i) {
        const long row = (long)(mBase + mt * 16 + hi * 8 + i);
        const long idx = row * 512 + d;
        const float v = acc[mt][nt][i] + bias;
        const float s = 1.0f / (1.0f + __expf(-v));
        gated[idx] = xf[idx] * s;    // gate applied to f32 x_freq
      }
    }
  }
}

// ---------------------------------------------------------------------------
extern "C" void kernel_launch(void* const* d_in, const int* in_sizes, int n_in,
                              void* d_out, int out_size, void* d_ws, size_t ws_size,
                              hipStream_t stream) {
  const float* x  = (const float*)d_in[0];  // (32,64,64,512)
  const float* g  = (const float*)d_in[1];  // (32,512)
  const float* Wg = (const float*)d_in[2];  // (512,512)
  const float* bg = (const float*)d_in[3];  // (512)
  const float* Wc = (const float*)d_in[4];  // (512,512)
  const float* bc = (const float*)d_in[5];  // (512)
  float* out = (float*)d_out;

  float*  ws   = (float*)d_ws;
  float*  D    = ws;                        // 4096 floats
  float*  Dt   = ws + 4096;                 // 4096
  float*  gmap = ws + 8192;                 // 16384
  __bf16* wcT  = (__bf16*)(ws + 24576);     // 262144 bf16 = 131072 float slots
  float*  buf1 = ws + 24576 + 131072;       // 67108864
  float*  buf2 = buf1 + 67108864;           // 67108864

  const long NSTRIDE_W = 32768L;            // W*C
  const long CSTRIDE   = 512L;              // C

  init_dct_kernel<<<16, 256, 0, stream>>>(D, Dt);
  convert_wc_kernel<<<1024, 256, 0, stream>>>(Wc, wcT);
  gate_gemm_kernel<<<64, 256, 0, stream>>>(g, Wg, bg, gmap);

  // 1) DCT along W: per (n,h): D @ X      (rows = w, stride 512)
  dct_gemm_kernel<<<2048, 256, 0, stream>>>(D, x, nullptr, buf1,
                                            /*jStride=*/NSTRIDE_W, /*rowStride=*/CSTRIDE);
  // 2) DCT along H: per (n,k2): D @ buf1  (rows = h, stride 32768)
  dct_gemm_kernel<<<2048, 256, 0, stream>>>(D, buf1, nullptr, buf2,
                                            /*jStride=*/CSTRIDE, /*rowStride=*/NSTRIDE_W);
  // 3) 1x1 conv + sigmoid gate + multiply (bf16 WMMA)
  conv_gate_kernel<<<dim3(2048, 2), 256, 0, stream>>>(buf2, wcT, bc, gmap, buf1);
  // 4) IDCT along H: per (n,k2): D^T @ buf1
  dct_gemm_kernel<<<2048, 256, 0, stream>>>(Dt, buf1, nullptr, buf2,
                                            /*jStride=*/CSTRIDE, /*rowStride=*/NSTRIDE_W);
  // 5) IDCT along W + residual: per (n,h): x + D^T @ buf2
  dct_gemm_kernel<<<2048, 256, 0, stream>>>(Dt, buf2, x, out,
                                            /*jStride=*/NSTRIDE_W, /*rowStride=*/CSTRIDE);
}
